// WarpingNetwork_50964081934441
// MI455X (gfx1250) — matrix operand order repaired and checked by reading, not compiled, not run on hardware
//
#include <hip/hip_runtime.h>
#include <hip/hip_bf16.h>

// ---------------------------------------------------------------------------
// Types
// ---------------------------------------------------------------------------
typedef _Float16 v16h __attribute__((ext_vector_type(16)));
typedef _Float16 v8h  __attribute__((ext_vector_type(8)));
typedef float    v8f  __attribute__((ext_vector_type(8)));
typedef unsigned int v4u __attribute__((ext_vector_type(4)));
typedef int      v4i  __attribute__((ext_vector_type(4)));
typedef int      v8i  __attribute__((ext_vector_type(8)));

#define LEAKY(v) ((v) >= 0.f ? (v) : 0.1f * (v))

constexpr int B_  = 2;
constexpr int C_  = 96;
constexpr int H_  = 256;
constexpr int W_  = 256;
constexpr int HW_ = H_ * W_;
constexpr int CHP1 = 136;  // conv1 input NHWC padded stride (97 used + pad)
constexpr int CHPM = 72;   // 64-ch NHWC padded stride

__device__ inline v16h ld16h(const _Float16* p) {
    union { v16h v; v8h h[2]; } u;
    u.h[0] = *(const v8h*)p;
    u.h[1] = *(const v8h*)(p + 8);
    return u.v;
}

__device__ inline v8f wmma_f16(v16h a, v16h b, v8f c) {
    return __builtin_amdgcn_wmma_f32_16x16x32_f16(false, a, false, b,
                                                  (short)0, c, false, false);
}

// ---------------------------------------------------------------------------
// TDM helper: 2D tile (nrows x rowElems of f16) global -> LDS.
// Builds the D# per CDNA5 ISA 8.3/8.4; issued by the calling lane's wave.
// ---------------------------------------------------------------------------
#if __has_builtin(__builtin_amdgcn_tensor_load_to_lds)
#define HAVE_TDM 1
#endif

__device__ inline void tdm_load_2d(const _Float16* gsrc, unsigned lds_off,
                                   unsigned rowElems, unsigned nrows,
                                   unsigned long long rowStrideElems) {
#if defined(HAVE_TDM)
    unsigned long long ga = (unsigned long long)(const void*)gsrc;
    v4u g0;
    g0[0] = 1u;                                   // count=1, user descriptor
    g0[1] = lds_off;                              // lds_addr (bytes)
    g0[2] = (unsigned)(ga & 0xffffffffu);         // global_addr[31:0]
    g0[3] = (unsigned)((ga >> 32) & 0x01ffffffu)  // global_addr[56:32]
          | 0x80000000u;                          // type=2 ("image")
    v8i g1;
    g1[0] = (1 << 16);                            // data_size=1 -> 2 bytes
    g1[1] = (int)((rowElems & 0xffffu) << 16);    // tensor_dim0[15:0]
    g1[2] = (int)(((rowElems >> 16) & 0xffffu) | ((nrows & 0xffffu) << 16));
    g1[3] = (int)(((nrows >> 16) & 0xffffu) | ((rowElems & 0xffffu) << 16)); // tile_dim0
    g1[4] = (int)(nrows & 0xffffu);               // tile_dim1 (tile_dim2=0)
    g1[5] = (int)(rowStrideElems & 0xffffffffu);  // tensor_dim0_stride[31:0]
    g1[6] = (int)((rowStrideElems >> 32) & 0xffffu);
    g1[7] = 0;
    v4i z4 = {};
#if __clang_major__ >= 23
    v8i z8 = {};
    __builtin_amdgcn_tensor_load_to_lds(g0, g1, z4, z4, z8, 0);
#else
    __builtin_amdgcn_tensor_load_to_lds(g0, g1, z4, z4, 0);
#endif
#else
    (void)gsrc; (void)lds_off; (void)rowElems; (void)nrows; (void)rowStrideElems;
#endif
}

__device__ inline void wait_tensorcnt0() {
#if __has_builtin(__builtin_amdgcn_s_wait_tensorcnt)
    __builtin_amdgcn_s_wait_tensorcnt(0);
#else
    asm volatile("s_wait_tensorcnt 0x0" ::: "memory");
#endif
}

// ---------------------------------------------------------------------------
// Kernel 1: backwarp + mean-abs metric; assemble conv1 input NHWC f16:
// xh[b][y][x][0]=metric, [1..96]=feature, [97..135]=0.
// ---------------------------------------------------------------------------
__global__ void warp_metric_kernel(const float* __restrict__ feat_src,
                                   const float* __restrict__ feat_oth,
                                   const float* __restrict__ bi_flow,
                                   int flowOff,
                                   _Float16* __restrict__ xh /* [B][H][W][CHP1] */) {
    int p = blockIdx.x * blockDim.x + threadIdx.x;
    if (p >= B_ * HW_) return;
    int b = p / HW_;
    int pix = p % HW_;
    int y = pix / W_, x = pix % W_;
    float fx = bi_flow[(size_t)(b * 4 + flowOff + 0) * HW_ + pix] * 0.125f;
    float fy = bi_flow[(size_t)(b * 4 + flowOff + 1) * HW_ + pix] * 0.125f;
    float tx = (float)x + fx, ty = (float)y + fy;
    float x0f = floorf(tx), y0f = floorf(ty);
    float ax = tx - x0f, ay = ty - y0f;
    int x0 = (int)x0f, y0 = (int)y0f;
    int xs[2] = {x0, x0 + 1}, ys[2] = {y0, y0 + 1};
    float wx[2] = {1.f - ax, ax}, wy[2] = {1.f - ay, ay};
    int idx[4];
    float wgt[4];
#pragma unroll
    for (int cy = 0; cy < 2; ++cy)
#pragma unroll
        for (int cx = 0; cx < 2; ++cx) {
            bool valid = xs[cx] >= 0 && xs[cx] < W_ && ys[cy] >= 0 && ys[cy] < H_;
            int xc = min(max(xs[cx], 0), W_ - 1);
            int yc = min(max(ys[cy], 0), H_ - 1);
            idx[cy * 2 + cx] = yc * W_ + xc;
            wgt[cy * 2 + cx] = valid ? wx[cx] * wy[cy] : 0.f;
        }
    const float* src = feat_src + (size_t)b * C_ * HW_;
    const float* oth = feat_oth + (size_t)b * C_ * HW_;
    _Float16* xo = xh + ((size_t)b * HW_ + pix) * CHP1;
    float acc = 0.f;
    for (int c = 0; c < C_; ++c) {
        const float* op = oth + (size_t)c * HW_;
        float s = wgt[0] * op[idx[0]] + wgt[1] * op[idx[1]] +
                  wgt[2] * op[idx[2]] + wgt[3] * op[idx[3]];
        float a = src[(size_t)c * HW_ + pix];
        acc += fabsf(a - s);
        xo[1 + c] = (_Float16)a;
    }
    xo[0] = (_Float16)(acc * (1.f / (float)C_));
#pragma unroll
    for (int c = 97; c < CHP1; ++c) xo[c] = (_Float16)0.f;
}

// ---------------------------------------------------------------------------
// Kernel 2: pre-swizzle conv weights (OIHW f32) into the contiguous-k A order that the B loads also use
// (16x32 f16 A: lanes0-15 M=0..15 K=0..15, lanes16-31 K=16..31).
// ---------------------------------------------------------------------------
__global__ void cvt_wfrag_kernel(const float* __restrict__ w, /* [64][Cin][3][3] */
                                 _Float16* __restrict__ dst, int Cin, int nChunk) {
    int tot = 4 * 9 * nChunk * 512;
    int e = blockIdx.x * blockDim.x + threadIdx.x;
    if (e >= tot) return;
    int f = e >> 9;
    int r = e & 511;
    int lane = r >> 4;
    int q = r & 15;
    int v = q >> 1, h = q & 1;
    int m = lane & 15, hi = lane >> 4;
    int k = hi * 16 + v * 2 + h;
    int chunk = f % nChunk;
    int t = (f / nChunk) % 9;
    int mtile = f / (nChunk * 9);
    int cin = chunk * 32 + k;
    int mg = mtile * 16 + m;
    int rr = t / 3, ss = t % 3;
    float val = (cin < Cin) ? w[((size_t)(mg * Cin + cin) * 3 + rr) * 3 + ss] : 0.f;
    dst[e] = (_Float16)val;
}

// ---------------------------------------------------------------------------
// Kernel 3: implicit-GEMM 3x3 conv, NHWC f16, v_wmma_f32_16x16x32_f16.
// Block = 256 thr (8 waves) -> 64 out-ch x 64 pixels on one row.
// Wave (mtile, half): A fragment reused across two 16-px B tiles.
// Input patch (3 rows x 66 x CHP f16, contiguous rows in NHWC) loaded by the
// Tensor Data Mover into LDS; edges pre-zeroed with ds stores.
// ---------------------------------------------------------------------------
template <int CHP, int NCHUNK, int CHPO>
__global__ void conv_wmma_kernel(const _Float16* __restrict__ xin,  /* [B][H][W][CHP] */
                                 const _Float16* __restrict__ wfrag,
                                 const float* __restrict__ bias,    /* [64] */
                                 _Float16* __restrict__ yout /* [B][H][W][CHPO] */) {
    constexpr int TW = 66;
    __shared__ _Float16 smem[3 * TW * CHP];
    int x0 = blockIdx.x * 64;
    int y = blockIdx.y;
    int b = blockIdx.z;
    const _Float16* xb = xin + (size_t)b * HW_ * CHP;

    // ---- pre-zero out-of-image tile regions (boundary blocks only) ----
    v8h zv = {};
    if (y == 0)
        for (int e = threadIdx.x; e < TW * CHP / 8; e += blockDim.x)
            *(v8h*)&smem[e * 8] = zv;
    if (y == H_ - 1)
        for (int e = threadIdx.x; e < TW * CHP / 8; e += blockDim.x)
            *(v8h*)&smem[2 * TW * CHP + e * 8] = zv;
    if (x0 == 0)
        for (int e = threadIdx.x; e < 3 * CHP / 8; e += blockDim.x) {
            int r = e / (CHP / 8), c8 = e % (CHP / 8);
            *(v8h*)&smem[(r * TW) * CHP + c8 * 8] = zv;
        }
    if (x0 + 64 == W_)
        for (int e = threadIdx.x; e < 3 * CHP / 8; e += blockDim.x) {
            int r = e / (CHP / 8), c8 = e % (CHP / 8);
            *(v8h*)&smem[(r * TW + TW - 1) * CHP + c8 * 8] = zv;
        }

    // ---- valid sub-rectangle geometry ----
    int gy0 = max(y - 1, 0);
    int nrows = min(y + 2, H_) - gy0;
    int r0 = gy0 - (y - 1);
    int gxs = max(x0 - 1, 0);
    int ncols = min(x0 + 65, W_) - gxs;
    int skipL = gxs - (x0 - 1);

#if defined(HAVE_TDM)
    if (threadIdx.x < 32) {
        unsigned lds_off = (unsigned)(size_t)&smem[(r0 * TW + skipL) * CHP];
        const _Float16* gsrc = xb + ((size_t)gy0 * W_ + gxs) * CHP;
        tdm_load_2d(gsrc, lds_off, (unsigned)(ncols * CHP), (unsigned)nrows,
                    (unsigned long long)W_ * CHP);
        wait_tensorcnt0();
    }
#else
    {
        int cpr = ncols * CHP / 8;  // 16B chunks per row
        for (int e = threadIdx.x; e < nrows * cpr; e += blockDim.x) {
            int rr = e / cpr, c8 = e % cpr;
            const _Float16* g = xb + ((size_t)(gy0 + rr) * W_ + gxs) * CHP + c8 * 8;
            _Float16* l = &smem[((r0 + rr) * TW + skipL) * CHP + c8 * 8];
            *(v8h*)l = *(const v8h*)g;
        }
    }
#endif
    __syncthreads();

    // ---- WMMA compute ----
    int wave = threadIdx.x >> 5;
    int lane = threadIdx.x & 31;
    int mtile = wave >> 1;
    int pxp = wave & 1;           // which 32-px half of the 64-px tile
    int n = lane & 15, khi = lane >> 4;

    v8f acc0 = {}, acc1 = {};
#pragma unroll 1
    for (int t = 0; t < 9; ++t) {
        int r = t / 3, s = t % 3;
        const _Float16* brow0 = &smem[(r * TW + pxp * 32 + n + s) * CHP];
        const _Float16* brow1 = brow0 + 16 * CHP;
#pragma unroll
        for (int chunk = 0; chunk < NCHUNK; ++chunk) {
            int fragIdx = (mtile * 9 + t) * NCHUNK + chunk;
            v16h a = ld16h(wfrag + (size_t)fragIdx * 512 + lane * 16);
            v16h b0 = ld16h(brow0 + chunk * 32 + khi * 16);
            v16h b1 = ld16h(brow1 + chunk * 32 + khi * 16);
            acc0 = wmma_f16(a, b0, acc0);  // A reused by both WMMAs
            acc1 = wmma_f16(a, b1, acc1);
        }
    }

    // ---- bias + leaky + contiguous 16B NHWC stores ----
    int m0 = mtile * 16 + 8 * khi;
    int xo0 = x0 + pxp * 32 + n;
    v8h o0, o1;
#pragma unroll
    for (int r8 = 0; r8 < 8; ++r8) {
        float v0 = acc0[r8] + bias[m0 + r8];
        float v1 = acc1[r8] + bias[m0 + r8];
        o0[r8] = (_Float16)LEAKY(v0);
        o1[r8] = (_Float16)LEAKY(v1);
    }
    size_t ob = ((size_t)b * HW_ + (size_t)y * W_ + xo0) * CHPO + m0;
    *(v8h*)&yout[ob] = o0;
    *(v8h*)&yout[ob + (size_t)16 * CHPO] = o1;
}

// ---------------------------------------------------------------------------
// Kernel 4: conv3 (64->1) + alpha scale + clip + exp (splat weight).
// ---------------------------------------------------------------------------
__global__ void conv3_metric_kernel(const _Float16* __restrict__ y2, /* [B][H][W][CHPM] */
                                    const float* __restrict__ w3,    /* [64*9] */
                                    const float* __restrict__ b3,
                                    const float* __restrict__ alpha,
                                    float* __restrict__ expm /* [B][H][W] */) {
    __shared__ float w3s[576];
    for (int i = threadIdx.x; i < 576; i += blockDim.x) w3s[i] = w3[i];
    __syncthreads();
    int p = blockIdx.x * blockDim.x + threadIdx.x;
    if (p >= B_ * HW_) return;
    int b = p / HW_, pix = p % HW_;
    int y = pix / W_, x = pix % W_;
    float s = b3[0];
    const _Float16* yb = y2 + (size_t)b * HW_ * CHPM;
#pragma unroll
    for (int t = 0; t < 9; ++t) {
        int gy = y + t / 3 - 1, gx = x + t % 3 - 1;
        if (gy < 0 || gy >= H_ || gx < 0 || gx >= W_) continue;
        const _Float16* cp = yb + ((size_t)gy * W_ + gx) * CHPM;
        for (int c = 0; c < 64; ++c) s += w3s[c * 9 + t] * (float)cp[c];
    }
    float met = fminf(fmaxf(alpha[0] * s, -20.f), 20.f);
    expm[p] = __expf(met);
}

// ---------------------------------------------------------------------------
// Kernel 5: zero splat accumulators.
// ---------------------------------------------------------------------------
__global__ void zero_kernel(float* __restrict__ p, size_t n) {
    size_t i = (size_t)blockIdx.x * blockDim.x + threadIdx.x;
    size_t stride = (size_t)gridDim.x * blockDim.x;
    for (; i < n; i += stride) p[i] = 0.f;
}

// ---------------------------------------------------------------------------
// Kernel 6: softmax-splat scatter (bilinear corners, exp(metric) weighted).
// ---------------------------------------------------------------------------
__global__ void splat_kernel(const float* __restrict__ feat,
                             const float* __restrict__ bi_flow, int flowOff,
                             const float* __restrict__ expm,
                             float* __restrict__ acc /* [B][97][H][W] */) {
    int p = blockIdx.x * blockDim.x + threadIdx.x;
    if (p >= B_ * HW_) return;
    int b = p / HW_, pix = p % HW_;
    int y = pix / W_, x = pix % W_;
    float fx = bi_flow[(size_t)(b * 4 + flowOff + 0) * HW_ + pix] * 0.125f;
    float fy = bi_flow[(size_t)(b * 4 + flowOff + 1) * HW_ + pix] * 0.125f;
    float tx = (float)x + fx, ty = (float)y + fy;
    float x0f = floorf(tx), y0f = floorf(ty);
    float ax = tx - x0f, ay = ty - y0f;
    int x0 = (int)x0f, y0 = (int)y0f;
    float w = expm[p];
    const float* fb = feat + (size_t)b * C_ * HW_;
    float* ab = acc + (size_t)b * 97 * HW_;
    int xs[2] = {x0, x0 + 1}, ys[2] = {y0, y0 + 1};
    float wx[2] = {1.f - ax, ax}, wy[2] = {1.f - ay, ay};
#pragma unroll
    for (int cy = 0; cy < 2; ++cy)
#pragma unroll
        for (int cx = 0; cx < 2; ++cx) {
            int xi = xs[cx], yi = ys[cy];
            if (xi < 0 || xi >= W_ || yi < 0 || yi >= H_) continue;
            float cw = wx[cx] * wy[cy] * w;
            if (cw == 0.f) continue;
            int di = yi * W_ + xi;
            for (int c = 0; c < C_; ++c)
                atomicAdd(&ab[(size_t)c * HW_ + di], fb[(size_t)c * HW_ + pix] * cw);
            atomicAdd(&ab[(size_t)96 * HW_ + di], cw);
        }
}

// ---------------------------------------------------------------------------
// Kernel 7: normalize splat: out = num / (den + 1e-7)
// ---------------------------------------------------------------------------
__global__ void normalize_kernel(const float* __restrict__ acc,
                                 float* __restrict__ out /* [B][96][H][W] */) {
    size_t i = (size_t)blockIdx.x * blockDim.x + threadIdx.x;
    if (i >= (size_t)B_ * C_ * HW_) return;
    size_t b = i / ((size_t)C_ * HW_);
    size_t rem = i % ((size_t)C_ * HW_);
    size_t pix = rem % HW_;
    float den = acc[(b * 97 + 96) * (size_t)HW_ + pix] + 1e-7f;
    out[i] = acc[b * 97 * (size_t)HW_ + rem] / den;
}

// ---------------------------------------------------------------------------
// Kernel 8: 81-displacement cost volume via WMMA band-GEMM.
// cost[i][dx] = band D[i][i+dx] of D = L_strip(16x96) * R_window(96x24),
// computed as two overlapping 16x16 WMMAs per K-chunk per dy.
// ---------------------------------------------------------------------------
__global__ void corr_wmma_kernel(const float* __restrict__ wl,
                                 const float* __restrict__ wr,
                                 float* __restrict__ cost /* [B][81][H][W] */) {
    constexpr int CHP2 = 104;
    __shared__ _Float16 sL[16 * CHP2];
    __shared__ _Float16 sR[9 * 24 * CHP2];
    __shared__ _Float16 sD[9 * 2 * 256];
    int x0 = blockIdx.x * 16;
    int y = blockIdx.y;
    int b = blockIdx.z;
    const float* lb = wl + (size_t)b * C_ * HW_;
    const float* rb = wr + (size_t)b * C_ * HW_;

    for (int e = threadIdx.x; e < 16 * C_; e += blockDim.x) {
        int i = e & 15, c = e >> 4;
        sL[i * CHP2 + c] = (_Float16)lb[(size_t)c * HW_ + y * W_ + x0 + i];
    }
    for (int e = threadIdx.x; e < 9 * 24 * C_; e += blockDim.x) {
        int j = e % 24;
        int c = (e / 24) % C_;
        int dyi = e / (24 * C_);
        int gy = y + dyi - 4;
        int gx = x0 - 4 + j;
        _Float16 val = (_Float16)0.f;
        if (gy >= 0 && gy < H_ && gx >= 0 && gx < W_)
            val = (_Float16)rb[(size_t)c * HW_ + gy * W_ + gx];
        sR[(dyi * 24 + j) * CHP2 + c] = val;
    }
    __syncthreads();

    int wave = threadIdx.x >> 5, lane = threadIdx.x & 31;
    int n = lane & 15, khi = lane >> 4;
    for (int dyi = wave; dyi < 9; dyi += 8) {  // wave0 also takes dy=8
        v8f d1 = {}, d2 = {};
#pragma unroll
        for (int chunk = 0; chunk < 3; ++chunk) {
            union { v16h v; v8h h[2]; } ua;  // A 16x32 f16 lane layout
            ua.h[0] = *(const v8h*)&sL[n * CHP2 + chunk * 32 + khi * 16];
            ua.h[1] = *(const v8h*)&sL[n * CHP2 + chunk * 32 + khi * 16 + 8];
            v16h b1 = ld16h(&sR[(dyi * 24 + n) * CHP2 + chunk * 32 + khi * 16]);
            v16h b2 = ld16h(&sR[(dyi * 24 + 8 + n) * CHP2 + chunk * 32 + khi * 16]);
            d1 = wmma_f16(ua.v, b1, d1);
            d2 = wmma_f16(ua.v, b2, d2);
        }
#pragma unroll
        for (int r8 = 0; r8 < 8; ++r8) {
            int m = r8 + 8 * khi;
            sD[(dyi * 2 + 0) * 256 + m * 16 + n] = (_Float16)d1[r8];
            sD[(dyi * 2 + 1) * 256 + m * 16 + n] = (_Float16)d2[r8];
        }
    }
    __syncthreads();

    for (int e = threadIdx.x; e < 81 * 16; e += blockDim.x) {
        int i = e & 15, d = e >> 4;
        int dy_ = d / 9, dx_ = d % 9;
        int j = i + dx_;
        float v = (j < 16) ? (float)sD[(dy_ * 2) * 256 + i * 16 + j]
                           : (float)sD[(dy_ * 2 + 1) * 256 + i * 16 + (j - 8)];
        v *= (1.f / 96.f);
        v = LEAKY(v);
        cost[((size_t)(b * 81 + d) * H_ + y) * W_ + x0 + i] = v;
    }
}

// ---------------------------------------------------------------------------
// Host launcher
// ---------------------------------------------------------------------------
extern "C" void kernel_launch(void* const* d_in, const int* in_sizes, int n_in,
                              void* d_out, int out_size, void* d_ws, size_t ws_size,
                              hipStream_t stream) {
    (void)in_sizes; (void)n_in; (void)out_size; (void)ws_size;
    const float* bi_flow = (const float*)d_in[0];
    const float* featL   = (const float*)d_in[1];
    const float* featR   = (const float*)d_in[2];
    const float* alpha   = (const float*)d_in[3];
    const float* w1      = (const float*)d_in[4];
    const float* b1      = (const float*)d_in[5];
    const float* w2      = (const float*)d_in[6];
    const float* b2      = (const float*)d_in[7];
    const float* w3      = (const float*)d_in[8];
    const float* b3      = (const float*)d_in[9];
    float* out = (float*)d_out;

    char* ws = (char*)d_ws;
    size_t off = 0;
    auto carve = [&](size_t bytes) -> char* {
        char* p = ws + off;
        off = (off + bytes + 255) & ~(size_t)255;
        return p;
    };
    _Float16* xhL    = (_Float16*)carve((size_t)B_ * HW_ * CHP1 * 2);
    _Float16* xhR    = (_Float16*)carve((size_t)B_ * HW_ * CHP1 * 2);
    _Float16* w1frag = (_Float16*)carve((size_t)4 * 9 * 4 * 512 * 2);
    _Float16* w2frag = (_Float16*)carve((size_t)4 * 9 * 2 * 512 * 2);
    _Float16* y1L    = (_Float16*)carve((size_t)B_ * HW_ * CHPM * 2);
    _Float16* y1R    = (_Float16*)carve((size_t)B_ * HW_ * CHPM * 2);
    _Float16* y2L    = (_Float16*)carve((size_t)B_ * HW_ * CHPM * 2);
    _Float16* y2R    = (_Float16*)carve((size_t)B_ * HW_ * CHPM * 2);
    float*    expmL  = (float*)carve((size_t)B_ * HW_ * 4);
    float*    expmR  = (float*)carve((size_t)B_ * HW_ * 4);
    float*    accAll = (float*)carve((size_t)2 * B_ * 97 * HW_ * 4);
    float*    accL   = accAll;
    float*    accR   = accAll + (size_t)B_ * 97 * HW_;

    float* outL = out;
    float* outR = out + (size_t)B_ * C_ * HW_;
    float* outC = out + (size_t)2 * B_ * C_ * HW_;

    const int T = 256;
    cvt_wfrag_kernel<<<(4 * 9 * 4 * 512 + T - 1) / T, T, 0, stream>>>(w1, w1frag, 97, 4);
    cvt_wfrag_kernel<<<(4 * 9 * 2 * 512 + T - 1) / T, T, 0, stream>>>(w2, w2frag, 64, 2);

    int pb = (B_ * HW_ + T - 1) / T;
    warp_metric_kernel<<<pb, T, 0, stream>>>(featL, featR, bi_flow, 0, xhL);
    warp_metric_kernel<<<pb, T, 0, stream>>>(featR, featL, bi_flow, 2, xhR);

    dim3 cgrid(W_ / 64, H_, B_);
    conv_wmma_kernel<CHP1, 4, CHPM><<<cgrid, T, 0, stream>>>(xhL, w1frag, b1, y1L);
    conv_wmma_kernel<CHP1, 4, CHPM><<<cgrid, T, 0, stream>>>(xhR, w1frag, b1, y1R);
    conv_wmma_kernel<CHPM, 2, CHPM><<<cgrid, T, 0, stream>>>(y1L, w2frag, b2, y2L);
    conv_wmma_kernel<CHPM, 2, CHPM><<<cgrid, T, 0, stream>>>(y1R, w2frag, b2, y2R);

    conv3_metric_kernel<<<pb, T, 0, stream>>>(y2L, w3, b3, alpha, expmL);
    conv3_metric_kernel<<<pb, T, 0, stream>>>(y2R, w3, b3, alpha, expmR);

    size_t nacc = (size_t)2 * B_ * 97 * HW_;
    zero_kernel<<<2048, T, 0, stream>>>(accAll, nacc);
    splat_kernel<<<pb, T, 0, stream>>>(featL, bi_flow, 0, expmL, accL);
    splat_kernel<<<pb, T, 0, stream>>>(featR, bi_flow, 2, expmR, accR);
    int nb = (int)(((size_t)B_ * C_ * HW_ + T - 1) / T);
    normalize_kernel<<<nb, T, 0, stream>>>(accL, outL);
    normalize_kernel<<<nb, T, 0, stream>>>(accR, outR);

    dim3 kgrid(W_ / 16, H_, B_);
    corr_wmma_kernel<<<kgrid, T, 0, stream>>>(outL, outR, outC);
}
